// MambaBlock_57844619543173
// MI455X (gfx1250) — matrix operand order repaired and checked by reading, not compiled, not run on hardware
//
#include <hip/hip_runtime.h>

typedef __attribute__((ext_vector_type(16))) __bf16 v16bf;
typedef __attribute__((ext_vector_type(8)))  float  v8f;

#define DEV __device__ __forceinline__

constexpr int BATCH = 4;
constexpr int DIMC  = 384;
constexpr int LSEQ  = 4096;
constexpr int M     = BATCH * LSEQ;     // 16384 tokens
constexpr int DI    = 768;              // d_inner
constexpr int NST   = 16;               // d_state
constexpr int MT    = M / 16;           // 1024 token tiles
constexpr int KS_IN = DIMC / 32;        // 12 k-steps (in_proj)
constexpr int KS_DI = DI / 32;          // 24 k-steps (x_proj/out_proj)
constexpr int CH    = 64;               // scan chunk length
constexpr int NC    = LSEQ / CH;        // 64 chunks per sequence
constexpr int NSEQ  = BATCH * DI;       // 3072 (b,d) sequences
constexpr int SPLANE= NSEQ * NST;       // 49152 scan lanes per chunk slab

// ---------------- workspace arena (bytes) ----------------
constexpr size_t SZ_XNA   = (size_t)MT * KS_IN * 512 * 2;      // xn A-pack bf16 (12.58MB)
constexpr size_t OFF_XNA  = 0;
constexpr size_t OFF_WIN  = OFF_XNA + SZ_XNA;                  // in_proj W pack (96 ntiles)
constexpr size_t SZ_WIN   = (size_t)(2 * DI / 16) * KS_IN * 512 * 2;
constexpr size_t OFF_WXP  = OFF_WIN + SZ_WIN;                  // x_proj W pack (64 cols pad)
constexpr size_t SZ_WXP   = (size_t)4 * KS_DI * 512 * 2;
constexpr size_t OFF_WDT  = OFF_WXP + SZ_WXP;                  // dt_proj W pack (K pad 32)
constexpr size_t SZ_WDT   = (size_t)(DI / 16) * 1 * 512 * 2;
constexpr size_t OFF_WOUT = OFF_WDT + SZ_WDT;                  // out_proj W pack
constexpr size_t SZ_WOUT  = (size_t)(DIMC / 16) * KS_DI * 512 * 2;
constexpr size_t OFF_UPRE = OFF_WOUT + SZ_WOUT;                // pre-conv u, f32 [M,768]
constexpr size_t SZ_F32MD = (size_t)M * DI * 4;
constexpr size_t OFF_ZS   = OFF_UPRE + SZ_F32MD;               // silu(z), f32 [M,768]
constexpr size_t OFF_U    = OFF_ZS + SZ_F32MD;                 // post-conv u, f32 [M,768]
constexpr size_t OFF_UA   = OFF_U + SZ_F32MD;                  // u A-pack bf16 (25.2MB)
constexpr size_t SZ_UA    = (size_t)MT * KS_DI * 512 * 2;
constexpr size_t OFF_XDBL = OFF_UA + SZ_UA;                    // x_dbl f32 [M,64]
constexpr size_t SZ_XDBL  = (size_t)M * 64 * 4;
constexpr size_t OFF_DTA  = OFF_XDBL + SZ_XDBL;                // dt A-pack (K=32) bf16
constexpr size_t SZ_DTA   = (size_t)MT * 512 * 2;
constexpr size_t OFF_DELTA= OFF_DTA + SZ_DTA;                  // delta f32 [M,768]
// aliases (producers dead before scan runs):
constexpr size_t OFF_YA   = OFF_UPRE;                          // y A-pack, aliases dead u_pre
constexpr size_t SZ_SCH   = (size_t)NC * SPLANE * 4;           // 12.58MB per slab
constexpr size_t OFF_SCH  = OFF_XNA;                           // S: aliases dead xnA (exact fit)
constexpr size_t OFF_PCH  = OFF_UA;                            // P: aliases dead uA
constexpr size_t OFF_H0   = OFF_UA + SZ_SCH;                   // h0: aliases dead uA (fits 25.2MB)

// --------- WMMA fragment element mapping (CDNA5 CDNA5 ISA) ---------
// A (16x32 bf16): lanes 0-15 M, lane group +16 for K bit3; j = (k>>4)*8 + (k&7)
DEV int frag_off_a(int m, int k) {
  return ((m & 15) + (((k >> 3) & 1) << 4)) * 16 + ((k >> 4) << 3) + (k & 7);
}
// B (32x16 bf16): N striped on lanes, lane group +16 for K bit3; j = (k>>4)*8 + (k&7)
DEV int frag_off_b(int n, int k) {
  return ((n & 15) + (((k >> 3) & 1) << 4)) * 16 + ((k >> 4) << 3) + (k & 7);
}

DEV v8f zero8() { v8f z = {0.f,0.f,0.f,0.f,0.f,0.f,0.f,0.f}; return z; }
DEV v8f wmma_bf16(v16bf a, v16bf b, v8f c) {
  return __builtin_amdgcn_wmma_f32_16x16x32_bf16(false, a, false, b, (short)0, c, false, false);
}
DEV float silu_f(float v) { return v / (1.f + __expf(-v)); }

// ================= weight packer: W[N][K] row-major -> B fragments =================
__global__ void __launch_bounds__(256) pack_b_kernel(const float* __restrict__ W,
                                                     __bf16* __restrict__ dst,
                                                     int N, int K, int Npad, int Kpad) {
  int idx = blockIdx.x * 256 + threadIdx.x;
  if (idx >= Npad * Kpad) return;
  int n = idx / Kpad;
  int k = idx % Kpad;
  float v = (n < N && k < K) ? W[(size_t)n * K + k] : 0.f;
  int ksteps = Kpad >> 5;
  size_t frag = (size_t)((n >> 4) * ksteps + (k >> 5)) * 512;
  dst[frag + frag_off_b(n, k & 31)] = (__bf16)v;
}

// ================= LayerNorm (+channel->token transpose) -> A-pack =================
__global__ void __launch_bounds__(256) ln_pack_kernel(const float* __restrict__ x,
                                                      const float* __restrict__ ln_w,
                                                      const float* __restrict__ ln_b,
                                                      __bf16* __restrict__ xnA) {
  __shared__ float tile[DIMC * 33];      // [c][tok] padded
  __shared__ float psum[256], psq[256];
  __shared__ float smu[32], srs[32];
  const int tid  = threadIdx.x;
  const int lane = tid & 31;
  const int wv   = tid >> 5;
  const int t0   = blockIdx.x * 32;      // 32 tokens, never crosses batch
  const int bb   = t0 >> 12;
  const int l0   = t0 & (LSEQ - 1);
  for (int c = wv; c < DIMC; c += 8)
    tile[c * 33 + lane] = x[((size_t)(bb * DIMC + c) << 12) + l0 + lane];
  __syncthreads();
  {
    const int tok = tid & 31, chunk = tid >> 5;
    float s = 0.f, q = 0.f;
    for (int i = 0; i < 48; ++i) {
      float v = tile[(chunk * 48 + i) * 33 + tok];
      s += v; q += v * v;
    }
    psum[chunk * 32 + tok] = s; psq[chunk * 32 + tok] = q;
  }
  __syncthreads();
  if (tid < 32) {
    float s = 0.f, q = 0.f;
    #pragma unroll
    for (int c = 0; c < 8; ++c) { s += psum[c * 32 + tid]; q += psq[c * 32 + tid]; }
    float mean = s * (1.f / DIMC);
    float var  = q * (1.f / DIMC) - mean * mean;
    smu[tid] = mean;
    srs[tid] = rsqrtf(var + 1e-5f);
  }
  __syncthreads();
  for (int e = tid; e < DIMC * 32; e += 256) {
    int c = e >> 5, tok = e & 31;
    float v = (tile[c * 33 + tok] - smu[tok]) * srs[tok] * ln_w[c] + ln_b[c];
    int t = t0 + tok;
    size_t frag = (size_t)((t >> 4) * KS_IN + (c >> 5)) * 512;
    xnA[frag + frag_off_a(t & 15, c & 31)] = (__bf16)v;
  }
}

// ================= in_proj GEMM: [M,384]x[384,1536] -> u_pre | silu(z) =================
__global__ void __launch_bounds__(256) gemm_in_proj(const __bf16* __restrict__ Ap,
                                                    const __bf16* __restrict__ Bp,
                                                    float* __restrict__ u_pre,
                                                    float* __restrict__ zs) {
  const int lane  = threadIdx.x & 31;
  const int wid   = (blockIdx.x * 256 + threadIdx.x) >> 5;
  const int mtile = wid & (MT - 1);
  const int ng    = wid >> 10;                      // 24 groups of 4 n-tiles
  const v16bf* ap = (const v16bf*)Ap + (size_t)mtile * KS_IN * 32 + lane;
  const v16bf* bp[4];
  v8f acc[4];
  #pragma unroll
  for (int t = 0; t < 4; ++t) {
    bp[t]  = (const v16bf*)Bp + (size_t)(ng * 4 + t) * KS_IN * 32 + lane;
    acc[t] = zero8();
  }
  for (int k = 0; k < KS_IN; ++k) {
    v16bf a = *ap; ap += 32;
    __builtin_prefetch((const void*)ap, 0, 0);
    #pragma unroll
    for (int t = 0; t < 4; ++t) {
      v16bf b = *bp[t]; bp[t] += 32;
      acc[t] = wmma_bf16(a, b, acc[t]);
    }
  }
  const int m0 = mtile * 16 + ((lane >> 4) << 3);
  #pragma unroll
  for (int t = 0; t < 4; ++t) {
    int n = ng * 64 + t * 16 + (lane & 15);
    #pragma unroll
    for (int r = 0; r < 8; ++r) {
      float v = acc[t][r];
      size_t m = (size_t)(m0 + r);
      if (n < DI) u_pre[m * DI + n] = v;
      else        zs[m * DI + (n - DI)] = silu_f(v);
    }
  }
}

// ================= causal depthwise conv3 + SiLU -> u (f32) + A-pack =================
__global__ void __launch_bounds__(256) conv_silu_pack(const float* __restrict__ u_pre,
                                                      const float* __restrict__ conv_w,
                                                      const float* __restrict__ conv_b,
                                                      float* __restrict__ u,
                                                      __bf16* __restrict__ uA) {
  size_t idx = (size_t)blockIdx.x * 256 + threadIdx.x;   // M*DI
  int d = (int)(idx % DI);
  size_t t = idx / DI;
  int l = (int)(t & (LSEQ - 1));
  float acc = conv_b[d] + conv_w[d * 3 + 2] * u_pre[idx];
  if (l >= 1) acc += conv_w[d * 3 + 1] * u_pre[idx - DI];
  if (l >= 2) acc += conv_w[d * 3 + 0] * u_pre[idx - 2 * DI];
  float uv = silu_f(acc);
  u[idx] = uv;
  size_t frag = ((t >> 4) * KS_DI + (size_t)(d >> 5)) * 512;
  uA[frag + frag_off_a((int)(t & 15), d & 31)] = (__bf16)uv;
}

// ================= x_proj GEMM: [M,768]x[768,64pad] -> x_dbl + dt A-pack =================
__global__ void __launch_bounds__(256) gemm_x_proj(const __bf16* __restrict__ Ap,
                                                   const __bf16* __restrict__ Bp,
                                                   float* __restrict__ xdbl,
                                                   __bf16* __restrict__ dtA) {
  const int lane  = threadIdx.x & 31;
  const int mtile = (blockIdx.x * 256 + threadIdx.x) >> 5;   // 1024 waves
  const v16bf* ap = (const v16bf*)Ap + (size_t)mtile * KS_DI * 32 + lane;
  const v16bf* bp[4];
  v8f acc[4];
  #pragma unroll
  for (int t = 0; t < 4; ++t) {
    bp[t]  = (const v16bf*)Bp + (size_t)t * KS_DI * 32 + lane;
    acc[t] = zero8();
  }
  for (int k = 0; k < KS_DI; ++k) {
    v16bf a = *ap; ap += 32;
    #pragma unroll
    for (int t = 0; t < 4; ++t) {
      v16bf b = *bp[t]; bp[t] += 32;
      acc[t] = wmma_bf16(a, b, acc[t]);
    }
  }
  const int m0 = mtile * 16 + ((lane >> 4) << 3);
  #pragma unroll
  for (int t = 0; t < 4; ++t) {
    int n = t * 16 + (lane & 15);
    #pragma unroll
    for (int r = 0; r < 8; ++r) {
      float v = acc[t][r];
      int m = m0 + r;
      if (n < 56) xdbl[(size_t)m * 64 + n] = v;
      if (n < 32)   // A-pack for dt GEMM (weight rows >=24 are zero, so pad cols harmless)
        dtA[(size_t)(m >> 4) * 512 + frag_off_a(m & 15, n)] = (__bf16)v;
    }
  }
}

// ================= dt GEMM: [M,32]x[32,768] + bias -> softplus -> delta =================
__global__ void __launch_bounds__(256) gemm_dt_proj(const __bf16* __restrict__ Ap,
                                                    const __bf16* __restrict__ Bp,
                                                    const float* __restrict__ dt_b,
                                                    float* __restrict__ delta) {
  const int lane  = threadIdx.x & 31;
  const int wid   = (blockIdx.x * 256 + threadIdx.x) >> 5;
  const int mtile = wid & (MT - 1);
  const int ng    = wid >> 10;                      // 12 groups of 4 n-tiles
  v16bf a = *((const v16bf*)Ap + (size_t)mtile * 32 + lane);
  const int m0 = mtile * 16 + ((lane >> 4) << 3);
  #pragma unroll
  for (int t = 0; t < 4; ++t) {
    v16bf b = *((const v16bf*)Bp + (size_t)(ng * 4 + t) * 32 + lane);
    v8f acc = wmma_bf16(a, b, zero8());
    int n = ng * 64 + t * 16 + (lane & 15);
    float bias = dt_b[n];
    #pragma unroll
    for (int r = 0; r < 8; ++r) {
      float v = acc[r] + bias;
      float sp = (v > 20.f) ? v : log1pf(__expf(v));
      delta[(size_t)(m0 + r) * DI + n] = sp;
    }
  }
}

// ========== chunked selective scan, phase A: per-chunk local state + coef product ==========
// grid: NC * 12 blocks; block = (batch, dgroup, chunk); lane-per-d, 16 states in VGPRs
__global__ void __launch_bounds__(256) scan_phaseA(const float* __restrict__ delta,
                                                   const float* __restrict__ u,
                                                   const float* __restrict__ xdbl,
                                                   const float* __restrict__ A_log,
                                                   float* __restrict__ S,
                                                   float* __restrict__ P) {
  const int tid = threadIdx.x;
  const int c   = blockIdx.x % NC;
  const int g   = blockIdx.x / NC;          // 0..11
  const int bb  = g / 3;
  const int d   = (g % 3) * 256 + tid;
  const int lane = tid & 31;
  float Ac[NST], h[NST], pA[NST];
  #pragma unroll
  for (int n = 0; n < NST; ++n) {
    Ac[n] = -__expf(A_log[d * NST + n]);
    h[n] = 0.f; pA[n] = 1.f;
  }
  const int t0 = (bb << 12) + c * CH;
  for (int s = 0; s < CH; ++s) {
    const int t = t0 + s;
    const size_t td = (size_t)t * DI + d;
    const float dt_ = delta[td];
    const float ut  = u[td];
    const float bc  = xdbl[(size_t)t * 64 + 24 + lane];   // lanes 0-15: B[n]
    const float du  = dt_ * ut;
    #pragma unroll
    for (int n = 0; n < NST; ++n) {
      float Bn = __shfl(bc, n, 32);
      float dA = __expf(dt_ * Ac[n]);
      pA[n] *= dA;
      h[n] = fmaf(dA, h[n], du * Bn);
    }
  }
  const size_t base = (size_t)c * SPLANE + ((size_t)(bb * DI + d) << 4);
  #pragma unroll
  for (int n = 0; n < NST; ++n) { S[base + n] = h[n]; P[base + n] = pA[n]; }
}

// ========== phase B: tiny serial scan over chunks, one thread per (b,d,n) ==========
__global__ void __launch_bounds__(256) scan_phaseB(const float* __restrict__ S,
                                                   const float* __restrict__ P,
                                                   float* __restrict__ h0) {
  const int i = blockIdx.x * 256 + threadIdx.x;   // 49152, coalesced chunk-major
  float carry = 0.f;
  for (int c = 0; c < NC; ++c) {
    const size_t o = (size_t)c * SPLANE + i;
    h0[o] = carry;
    carry = fmaf(P[o], carry, S[o]);
  }
}

// ========== phase C: re-run chunk from corrected h0, gate, pack y for out_proj ==========
__global__ void __launch_bounds__(256) scan_phaseC(const float* __restrict__ delta,
                                                   const float* __restrict__ u,
                                                   const float* __restrict__ xdbl,
                                                   const float* __restrict__ zs,
                                                   const float* __restrict__ A_log,
                                                   const float* __restrict__ d_skip,
                                                   const float* __restrict__ h0,
                                                   __bf16* __restrict__ yA) {
  const int tid = threadIdx.x;
  const int c   = blockIdx.x % NC;
  const int g   = blockIdx.x / NC;
  const int bb  = g / 3;
  const int d   = (g % 3) * 256 + tid;
  const int lane = tid & 31;
  float Ac[NST], h[NST];
  const size_t base = (size_t)c * SPLANE + ((size_t)(bb * DI + d) << 4);
  #pragma unroll
  for (int n = 0; n < NST; ++n) {
    Ac[n] = -__expf(A_log[d * NST + n]);
    h[n] = h0[base + n];
  }
  const float skip = d_skip[d];
  const int kk = d & 31;
  const int lp = ((kk >> 3) & 1) << 4;              // lane-group offset for A-pack
  const int jp = ((kk >> 4) << 3) + (kk & 7);       // element offset for A-pack
  const int kstep = d >> 5;
  const int t0 = (bb << 12) + c * CH;
  for (int s = 0; s < CH; ++s) {
    const int t = t0 + s;
    const size_t td = (size_t)t * DI + d;
    const float dt_ = delta[td];
    const float ut  = u[td];
    const float bc  = xdbl[(size_t)t * 64 + 24 + lane];   // 0-15: B[n], 16-31: C[n]
    const float du  = dt_ * ut;
    float y = 0.f;
    #pragma unroll
    for (int n = 0; n < NST; ++n) {
      float Bn = __shfl(bc, n, 32);
      float Cn = __shfl(bc, 16 + n, 32);
      float dA = __expf(dt_ * Ac[n]);
      h[n] = fmaf(dA, h[n], du * Bn);
      y    = fmaf(h[n], Cn, y);
    }
    float yf = (y + ut * skip) * zs[td];
    size_t frag = ((size_t)(t >> 4) * KS_DI + kstep) * 512;
    yA[frag + (size_t)(((t & 15) + lp) << 4) + jp] = (__bf16)yf;
  }
}

// ================= out_proj GEMM + residual, LDS-transposed coalesced store =================
__global__ void __launch_bounds__(256) gemm_out_proj(const __bf16* __restrict__ Ap,
                                                     const __bf16* __restrict__ Bp,
                                                     const float* __restrict__ xres,
                                                     float* __restrict__ out) {
  __shared__ float tile[8][16 * 17];
  const int lane  = threadIdx.x & 31;
  const int wv    = threadIdx.x >> 5;
  const int wid   = (blockIdx.x * 256 + threadIdx.x) >> 5;
  const int mtile = wid & (MT - 1);
  const int ng    = wid >> 10;                      // 6 groups of 4 n-tiles (N=384)
  const v16bf* ap = (const v16bf*)Ap + (size_t)mtile * KS_DI * 32 + lane;
  const v16bf* bp[4];
  v8f acc[4];
  #pragma unroll
  for (int t = 0; t < 4; ++t) {
    bp[t]  = (const v16bf*)Bp + (size_t)(ng * 4 + t) * KS_DI * 32 + lane;
    acc[t] = zero8();
  }
  for (int k = 0; k < KS_DI; ++k) {
    v16bf a = *ap; ap += 32;
    __builtin_prefetch((const void*)ap, 0, 0);
    #pragma unroll
    for (int t = 0; t < 4; ++t) {
      v16bf b = *bp[t]; bp[t] += 32;
      acc[t] = wmma_bf16(a, b, acc[t]);
    }
  }
  for (int t = 0; t < 4; ++t) {
    #pragma unroll
    for (int r = 0; r < 8; ++r) {
      int ml = r + ((lane >> 4) << 3);
      tile[wv][ml * 17 + (lane & 15)] = acc[t][r];
    }
    __syncthreads();
    #pragma unroll
    for (int r = 0; r < 8; ++r) {
      int idx = r * 32 + lane;
      int ml = idx & 15, nl = idx >> 4;
      float v = tile[wv][ml * 17 + nl];
      int token = mtile * 16 + ml;
      int b = token >> 12, l = token & (LSEQ - 1);
      int c = ng * 64 + t * 16 + nl;
      size_t o = ((size_t)(b * DIMC + c) << 12) + l;
      out[o] = xres[o] + v;
    }
    __syncthreads();
  }
}

// ================================ launcher ================================
extern "C" void kernel_launch(void* const* d_in, const int* in_sizes, int n_in,
                              void* d_out, int out_size, void* d_ws, size_t ws_size,
                              hipStream_t stream) {
  (void)in_sizes; (void)n_in; (void)out_size; (void)ws_size;
  const float* x      = (const float*)d_in[0];
  const float* ln_w   = (const float*)d_in[1];
  const float* ln_b   = (const float*)d_in[2];
  const float* in_w   = (const float*)d_in[3];
  const float* conv_w = (const float*)d_in[4];
  const float* conv_b = (const float*)d_in[5];
  const float* xp_w   = (const float*)d_in[6];
  const float* dtp_w  = (const float*)d_in[7];
  const float* dtp_b  = (const float*)d_in[8];
  const float* A_log  = (const float*)d_in[9];
  const float* dskip  = (const float*)d_in[10];
  const float* out_w  = (const float*)d_in[11];
  char* ws = (char*)d_ws;
  float* out = (float*)d_out;

  __bf16* xnA  = (__bf16*)(ws + OFF_XNA);
  __bf16* win  = (__bf16*)(ws + OFF_WIN);
  __bf16* wxp  = (__bf16*)(ws + OFF_WXP);
  __bf16* wdt  = (__bf16*)(ws + OFF_WDT);
  __bf16* wout = (__bf16*)(ws + OFF_WOUT);
  float*  upre = (float*)(ws + OFF_UPRE);
  float*  zsb  = (float*)(ws + OFF_ZS);
  float*  ub   = (float*)(ws + OFF_U);
  __bf16* uA   = (__bf16*)(ws + OFF_UA);
  float*  xdbl = (float*)(ws + OFF_XDBL);
  __bf16* dtA  = (__bf16*)(ws + OFF_DTA);
  float*  delta= (float*)(ws + OFF_DELTA);
  __bf16* yA   = (__bf16*)(ws + OFF_YA);   // aliases dead u_pre
  float*  Sch  = (float*)(ws + OFF_SCH);   // aliases dead xnA
  float*  Pch  = (float*)(ws + OFF_PCH);   // aliases dead uA
  float*  h0b  = (float*)(ws + OFF_H0);    // aliases dead uA (+12.58MB)

  // weight packing (B-fragment layout, zero-padded)
  pack_b_kernel<<<(1536 * 384 + 255) / 256, 256, 0, stream>>>(in_w,  win,  1536, 384, 1536, 384);
  pack_b_kernel<<<(64 * 768 + 255) / 256,   256, 0, stream>>>(xp_w,  wxp,  56,   768, 64,   768);
  pack_b_kernel<<<(768 * 32 + 255) / 256,   256, 0, stream>>>(dtp_w, wdt,  768,  24,  768,  32);
  pack_b_kernel<<<(384 * 768 + 255) / 256,  256, 0, stream>>>(out_w, wout, 384,  768, 384,  768);

  ln_pack_kernel<<<M / 32, 256, 0, stream>>>(x, ln_w, ln_b, xnA);
  gemm_in_proj<<<(MT * 24) / 8, 256, 0, stream>>>(xnA, win, upre, zsb);
  conv_silu_pack<<<(int)(((size_t)M * DI) / 256), 256, 0, stream>>>(upre, conv_w, conv_b, ub, uA);
  gemm_x_proj<<<MT / 8, 256, 0, stream>>>(uA, wxp, xdbl, dtA);
  gemm_dt_proj<<<(MT * 12) / 8, 256, 0, stream>>>(dtA, wdt, dtp_b, delta);
  // chunked parallel scan: 768-block phases instead of a 12-block serial scan
  scan_phaseA<<<NC * 12, 256, 0, stream>>>(delta, ub, xdbl, A_log, Sch, Pch);
  scan_phaseB<<<SPLANE / 256, 256, 0, stream>>>(Sch, Pch, h0b);
  scan_phaseC<<<NC * 12, 256, 0, stream>>>(delta, ub, xdbl, zsb, A_log, dskip, h0b, yA);
  gemm_out_proj<<<(MT * 6) / 8, 256, 0, stream>>>(yA, wout, x, out);
}
